// EntropyEvaluator_87832081203316
// MI455X (gfx1250) — matrix operand order repaired and checked by reading, not compiled, not run on hardware
//
#include <hip/hip_runtime.h>
#include <hip/hip_bf16.h>

// ---------------------------------------------------------------------------
// GNN forward (MLP encoder -> 2x GCNConv -> classifier) for MI455X / gfx1250.
// Dense GEMMs: v_wmma_f32_16x16x32_bf16, fp32 accumulate. Weights are
// pre-converted once to bf16 column-major so every B-fragment is two b128
// loads; per k-step ALL B-fragments are fetched into distinct registers first
// (one wide s_clause) and then drained through back-to-back WMMAs. Edge
// aggregation uses wave-uniform (scalarized) indices + f32 global atomics.
// ---------------------------------------------------------------------------

typedef __bf16 bf16_t;
typedef __attribute__((ext_vector_type(16))) __bf16 v16bf;
typedef __attribute__((ext_vector_type(8)))  __bf16 v8bf;
typedef __attribute__((ext_vector_type(8)))  float  v8f;
typedef __attribute__((ext_vector_type(4)))  float  f4;

#define LRELU_SLOPE 0.01f

__device__ __forceinline__ float lrelu(float v) { return v >= 0.f ? v : LRELU_SLOPE * v; }

__device__ __forceinline__ v8f wmma_bf16(v16bf a, v16bf b, v8f c) {
  // (neg_a, A, neg_b, B, c_mod, C, reuse_a, reuse_b)
  return __builtin_amdgcn_wmma_f32_16x16x32_bf16(false, a, false, b, (short)0, c, false, false);
}

// WMMA VGPR layouts (wave32):
//  A (16x32 bf16): lane l -> row M=l&15; elem j -> K = kb + 8*(l>>4) + j + ((j>>3)<<3)
//                  => per lane two contiguous 8-element K-runs.
//  B (32x16 bf16): lane l -> col N=l&15; elem j -> K = kb + 8*(l>>4) + j + ((j>>3)<<3)
//                  => per lane two contiguous 8-element K-runs (two b128 loads
//                     from a column-major bf16 weight image).
//  C/D (16x16 f32): lane l, vgpr r -> (M = r + 8*(l>>4), N = l&15).

// Load A-fragment from an fp32 row (global), converting to bf16.
__device__ __forceinline__ v16bf load_afrag_f32(const float* __restrict__ row, int ak) {
  const f4 a0 = *(const f4*)(row + ak);
  const f4 a1 = *(const f4*)(row + ak + 4);
  const f4 a2 = *(const f4*)(row + ak + 16);
  const f4 a3 = *(const f4*)(row + ak + 20);
  v16bf a;
#pragma unroll
  for (int j = 0; j < 4; ++j) {
    a[j]      = (bf16_t)a0[j];
    a[j + 4]  = (bf16_t)a1[j];
    a[j + 8]  = (bf16_t)a2[j];
    a[j + 12] = (bf16_t)a3[j];
  }
  return a;
}

// Load A-fragment from a bf16 row (LDS).
__device__ __forceinline__ v16bf load_afrag_bf16(const bf16_t* row, int ak) {
  const v8bf lo = *(const v8bf*)(row + ak);
  const v8bf hi = *(const v8bf*)(row + ak + 16);
  v16bf a;
#pragma unroll
  for (int j = 0; j < 8; ++j) { a[j] = lo[j]; a[j + 8] = hi[j]; }
  return a;
}

// Load B-fragment from a column-major bf16 weight image (leading dim = K).
__device__ __forceinline__ v16bf load_bfrag(const bf16_t* __restrict__ Bt, int n, int K, int k0) {
  const bf16_t* p = Bt + (size_t)n * K + k0;
  const v8bf lo = *(const v8bf*)(p);
  const v8bf hi = *(const v8bf*)(p + 16);
  v16bf b;
#pragma unroll
  for (int j = 0; j < 8; ++j) { b[j] = lo[j]; b[j + 8] = hi[j]; }
  return b;
}

// ---------------- weight pre-conversion: fp32 row-major -> bf16 col-major ---
// W1:512x128 W2:128x64 Wg1:64x64 Wg2:64x64
__global__ void prep_weights_kernel(const float* __restrict__ W1, const float* __restrict__ W2,
                                    const float* __restrict__ Wg1, const float* __restrict__ Wg2,
                                    bf16_t* __restrict__ W1t, bf16_t* __restrict__ W2t,
                                    bf16_t* __restrict__ Wg1t, bf16_t* __restrict__ Wg2t)
{
  int t = blockIdx.x * blockDim.x + threadIdx.x;
  if (t < 512 * 128) {
    int k = t >> 7, n = t & 127;                       // W1[k][n]
    W1t[(size_t)n * 512 + k] = (bf16_t)W1[t];
  } else if (t < 512 * 128 + 128 * 64) {
    int u = t - 512 * 128;
    int k = u >> 6, n = u & 63;                        // W2[k][n]
    W2t[(size_t)n * 128 + k] = (bf16_t)W2[u];
  } else if (t < 512 * 128 + 128 * 64 + 64 * 64) {
    int u = t - (512 * 128 + 128 * 64);
    int k = u >> 6, n = u & 63;
    Wg1t[(size_t)n * 64 + k] = (bf16_t)Wg1[u];
  } else if (t < 512 * 128 + 128 * 64 + 2 * 64 * 64) {
    int u = t - (512 * 128 + 128 * 64 + 64 * 64);
    int k = u >> 6, n = u & 63;
    Wg2t[(size_t)n * 64 + k] = (bf16_t)Wg2[u];
  }
}

// --------------------------- fused MLP encoder -----------------------------
// h2[i,0:64] = lrelu(lrelu(x[i,:] @ W1 + b1) @ W2 + b2);  one wave = 16 rows.
__global__ __launch_bounds__(32)
void encode_kernel(const float* __restrict__ x,
                   const bf16_t* __restrict__ W1t, const float* __restrict__ b1,
                   const bf16_t* __restrict__ W2t, const float* __restrict__ b2,
                   float* __restrict__ hout, int N)
{
  __shared__ bf16_t lds_h1[16 * 128];

  const int l     = threadIdx.x;
  const int m     = l & 15;
  const int khalf = l >> 4;
  const int node0 = blockIdx.x * 16;
  const int row   = node0 + m;
  const bool rok  = row < N;
  const float* xrow = x + (size_t)(rok ? row : 0) * 512;

  v8f acc[8];
#pragma unroll
  for (int t = 0; t < 8; ++t)
#pragma unroll
    for (int e = 0; e < 8; ++e) acc[t][e] = 0.f;

  // ---- GEMM1: [16x512] @ [512x128], K-steps of 32 ----
  for (int kb = 0; kb < 512; kb += 32) {
    const v16bf a = load_afrag_f32(xrow, kb + khalf * 8);
    const int k0 = kb + khalf * 8;
    v16bf bfr[8];
#pragma unroll
    for (int nt = 0; nt < 8; ++nt)
      bfr[nt] = load_bfrag(W1t, nt * 16 + m, 512, k0);
#pragma unroll
    for (int nt = 0; nt < 8; ++nt)
      acc[nt] = wmma_bf16(a, bfr[nt], acc[nt]);
  }

  // bias + lrelu, restripe D-layout -> row-major bf16 tile in LDS
#pragma unroll
  for (int nt = 0; nt < 8; ++nt) {
    const int n = nt * 16 + m;
#pragma unroll
    for (int r = 0; r < 8; ++r) {
      const int mr = r + 8 * khalf;
      lds_h1[mr * 128 + n] = (bf16_t)lrelu(acc[nt][r] + b1[n]);
    }
  }
  __syncthreads();

  // ---- GEMM2: [16x128] @ [128x64] ----
  v8f acc2[4];
#pragma unroll
  for (int t = 0; t < 4; ++t)
#pragma unroll
    for (int e = 0; e < 8; ++e) acc2[t][e] = 0.f;

#pragma unroll
  for (int kb = 0; kb < 128; kb += 32) {
    const v16bf a = load_afrag_bf16(&lds_h1[m * 128], kb + khalf * 8);
    const int k0 = kb + khalf * 8;
    v16bf bfr[4];
#pragma unroll
    for (int nt = 0; nt < 4; ++nt)
      bfr[nt] = load_bfrag(W2t, nt * 16 + m, 128, k0);
#pragma unroll
    for (int nt = 0; nt < 4; ++nt)
      acc2[nt] = wmma_bf16(a, bfr[nt], acc2[nt]);
  }

#pragma unroll
  for (int nt = 0; nt < 4; ++nt) {
    const int n = nt * 16 + m;
#pragma unroll
    for (int r = 0; r < 8; ++r) {
      const int orow = node0 + r + 8 * khalf;
      if (orow < N)
        hout[(size_t)orow * 64 + n] = lrelu(acc2[nt][r] + b2[n]);
    }
  }
}

// --------------------------- 64x64 GCN weight GEMM -------------------------
__global__ __launch_bounds__(32)
void gemm64_kernel(const float* __restrict__ h, const bf16_t* __restrict__ Wgt,
                   float* __restrict__ hw, int N)
{
  const int l     = threadIdx.x;
  const int m     = l & 15;
  const int khalf = l >> 4;
  const int node0 = blockIdx.x * 16;
  const int row   = node0 + m;
  const bool rok  = row < N;
  const float* hrow = h + (size_t)(rok ? row : 0) * 64;

  v8f acc[4];
#pragma unroll
  for (int t = 0; t < 4; ++t)
#pragma unroll
    for (int e = 0; e < 8; ++e) acc[t][e] = 0.f;

#pragma unroll
  for (int kb = 0; kb < 64; kb += 32) {
    const v16bf a = load_afrag_f32(hrow, kb + khalf * 8);
    const int k0 = kb + khalf * 8;
    v16bf bfr[4];
#pragma unroll
    for (int nt = 0; nt < 4; ++nt)
      bfr[nt] = load_bfrag(Wgt, nt * 16 + m, 64, k0);
#pragma unroll
    for (int nt = 0; nt < 4; ++nt)
      acc[nt] = wmma_bf16(a, bfr[nt], acc[nt]);
  }

#pragma unroll
  for (int nt = 0; nt < 4; ++nt) {
    const int n = nt * 16 + m;
#pragma unroll
    for (int r = 0; r < 8; ++r) {
      const int orow = node0 + r + 8 * khalf;
      if (orow < N)
        hw[(size_t)orow * 64 + n] = acc[nt][r];
    }
  }
}

// --------------------------- degree / norm ---------------------------------
__global__ void deg_init_kernel(float* __restrict__ deg, int N) {
  int i = blockIdx.x * blockDim.x + threadIdx.x;
  if (i < N) deg[i] = 1.0f;  // self-loop
}
__global__ void deg_accum_kernel(const long long* __restrict__ ei, float* __restrict__ deg, int E) {
  int e = blockIdx.x * blockDim.x + threadIdx.x;
  if (e < E) atomicAdd(&deg[(int)ei[(size_t)E + e]], 1.0f);
}
__global__ void deg_finalize_kernel(float* __restrict__ deg, int N) {
  int i = blockIdx.x * blockDim.x + threadIdx.x;
  if (i < N) deg[i] = rsqrtf(deg[i]);  // store dinv in place
}

// --------------------------- edge aggregation ------------------------------
__global__ void zero4_kernel(float* __restrict__ p, int n4) {
  int i = blockIdx.x * blockDim.x + threadIdx.x;
  if (i < n4) { f4 z; z[0] = z[1] = z[2] = z[3] = 0.f; *(f4*)(p + (size_t)i * 4) = z; }
}
// 64 consecutive threads share one edge -> edge index is wave-uniform:
// scalarize so the int64 index loads + coefficient live in SGPRs (s_load).
__global__ void scatter_kernel(const long long* __restrict__ ei, const float* __restrict__ dinv,
                               const float* __restrict__ hw, float* __restrict__ agg, int E)
{
  long long t = (long long)blockIdx.x * blockDim.x + threadIdx.x;
  int e = __builtin_amdgcn_readfirstlane((int)(t >> 6));
  int c = (int)(t & 63);
  if (e < E) {
    int s = (int)ei[e];
    int d = (int)ei[(size_t)E + e];
    float coef = dinv[s] * dinv[d];
    atomicAdd(&agg[(size_t)d * 64 + c], hw[(size_t)s * 64 + c] * coef);
  }
}
// one thread per (node, 4 channels)
__global__ void gcn_finalize_kernel(const float* __restrict__ agg, const float* __restrict__ hw,
                                    const float* __restrict__ dinv, const float* __restrict__ b,
                                    float* __restrict__ out, int N)
{
  int t = blockIdx.x * blockDim.x + threadIdx.x;
  int i = t >> 4, c4 = (t & 15) * 4;
  if (i < N) {
    float di = dinv[i];
    float dii = di * di;
    const f4 a = *(const f4*)(agg + (size_t)i * 64 + c4);
    const f4 w = *(const f4*)(hw + (size_t)i * 64 + c4);
    const f4 bb = *(const f4*)(b + c4);
    f4 o;
#pragma unroll
    for (int j = 0; j < 4; ++j) o[j] = lrelu(a[j] + w[j] * dii + bb[j]);
    *(f4*)(out + (size_t)i * 64 + c4) = o;
  }
}

// --------------------------- classifier ------------------------------------
__global__ void classifier_kernel(const float* __restrict__ h, const float* __restrict__ Wc,
                                  const float* __restrict__ bc, float* __restrict__ logits, int N)
{
  int t = blockIdx.x * blockDim.x + threadIdx.x;
  int i = t >> 1, c = t & 1;
  if (i < N) {
    float s = bc[c];
#pragma unroll
    for (int k4 = 0; k4 < 16; ++k4) {
      const f4 hv = *(const f4*)(h + (size_t)i * 64 + k4 * 4);
#pragma unroll
      for (int j = 0; j < 4; ++j) s += hv[j] * Wc[(k4 * 4 + j) * 2 + c];
    }
    logits[(size_t)i * 2 + c] = s;
  }
}

// ---------------------------------------------------------------------------
extern "C" void kernel_launch(void* const* d_in, const int* in_sizes, int n_in,
                              void* d_out, int out_size, void* d_ws, size_t ws_size,
                              hipStream_t stream)
{
  (void)n_in; (void)out_size; (void)ws_size;
  const float* x   = (const float*)d_in[0];
  const float* W1  = (const float*)d_in[1];
  const float* b1  = (const float*)d_in[2];
  const float* W2  = (const float*)d_in[3];
  const float* b2  = (const float*)d_in[4];
  const float* Wg1 = (const float*)d_in[5];
  const float* bg1 = (const float*)d_in[6];
  const float* Wg2 = (const float*)d_in[7];
  const float* bg2 = (const float*)d_in[8];
  const float* Wc  = (const float*)d_in[9];
  const float* bc  = (const float*)d_in[10];
  const long long* ei = (const long long*)d_in[11];  // int64 edge_index [2,E]

  const int N = in_sizes[0] / 512;
  const int E = in_sizes[11] / 2;

  float* ws   = (float*)d_ws;
  float* dinv = ws;                               // N
  float* h2   = dinv + N;                         // N*64  (encoder out / layer-1 out)
  float* hw   = h2 + (size_t)N * 64;              // N*64
  float* agg  = hw + (size_t)N * 64;              // N*64
  bf16_t* W1t  = (bf16_t*)(agg + (size_t)N * 64); // 128*512 bf16, col-major
  bf16_t* W2t  = W1t + 512 * 128;                 // 64*128
  bf16_t* Wg1t = W2t + 128 * 64;                  // 64*64
  bf16_t* Wg2t = Wg1t + 64 * 64;                  // 64*64

  float* logits = (float*)d_out;                  // [N,2]
  float* hout   = logits + (size_t)2 * N;         // [N,64]

  const int T = 256;
  const int WTOT = 512 * 128 + 128 * 64 + 2 * 64 * 64;

  // weights -> bf16 col-major (tiny, one-time per launch)
  prep_weights_kernel<<<(WTOT + T - 1) / T, T, 0, stream>>>(W1, W2, Wg1, Wg2, W1t, W2t, Wg1t, Wg2t);

  // degrees -> dinv
  deg_init_kernel<<<(N + T - 1) / T, T, 0, stream>>>(dinv, N);
  deg_accum_kernel<<<(E + T - 1) / T, T, 0, stream>>>(ei, dinv, E);
  deg_finalize_kernel<<<(N + T - 1) / T, T, 0, stream>>>(dinv, N);

  // MLP encoder
  encode_kernel<<<(N + 15) / 16, 32, 0, stream>>>(x, W1t, b1, W2t, b2, h2, N);

  const unsigned ncBlocks   = (unsigned)(((size_t)N * 16 + T - 1) / T);   // finalize: N*16 threads
  const unsigned zBlocks    = (unsigned)(((size_t)N * 16 + T - 1) / T);   // zero: N*64/4 threads
  const unsigned edgeBlocks = (unsigned)(((size_t)E * 64 + T - 1) / T);

  // GCN layer 1 (output overwrites h2)
  gemm64_kernel<<<(N + 15) / 16, 32, 0, stream>>>(h2, Wg1t, hw, N);
  zero4_kernel<<<zBlocks, T, 0, stream>>>(agg, N * 16);
  scatter_kernel<<<edgeBlocks, T, 0, stream>>>(ei, dinv, hw, agg, E);
  gcn_finalize_kernel<<<ncBlocks, T, 0, stream>>>(agg, hw, dinv, bg1, h2, N);

  // GCN layer 2 (output goes straight to d_out's h region)
  gemm64_kernel<<<(N + 15) / 16, 32, 0, stream>>>(h2, Wg2t, hw, N);
  zero4_kernel<<<zBlocks, T, 0, stream>>>(agg, N * 16);
  scatter_kernel<<<edgeBlocks, T, 0, stream>>>(ei, dinv, hw, agg, E);
  gcn_finalize_kernel<<<ncBlocks, T, 0, stream>>>(agg, hw, dinv, bg2, hout, N);

  // classifier
  classifier_kernel<<<(unsigned)(((size_t)2 * N + T - 1) / T), T, 0, stream>>>(hout, Wc, bc, logits, N);
}
